// AttentionSubsample_46901042872661
// MI455X (gfx1250) — matrix operand order repaired and checked by reading, not compiled, not run on hardware
//
#include <hip/hip_runtime.h>

typedef __attribute__((ext_vector_type(16))) _Float16 v16h;
typedef __attribute__((ext_vector_type(8)))  float    v8f;
typedef __attribute__((ext_vector_type(4)))  unsigned v4u;
typedef __attribute__((ext_vector_type(4)))  float    v4f;

#define B_SZ    32
#define N_TOK   784
#define NQ      196
#define NQ_PAD  208
#define IN_DIMC 384
#define H_HEADS 16
#define KD      32
#define DVDIM   64
#define DHD     1024
#define HIDC    1536
#define NHKD    512
#define OUTD    512
#define NKP     800          // keys padded to 25*32
#define SCALEQK 0.17677669529663687f
#define EPSBN   1e-5f

union FragU { v4u u[2]; v16h h; };

__device__ __forceinline__ unsigned pack2h(float a, float b) {
  union { _Float16 h[2]; unsigned u; } P;
  P.h[0] = (_Float16)a; P.h[1] = (_Float16)b;
  return P.u;
}

// A fragment 16x32 f16 (MxK) from LDS stored as u32 k-pairs, row stride rs dwords.
// Lane L<16 holds row M=L, K 0..15; lane L>=16 holds K 16..31.
__device__ __forceinline__ v16h frag_a_lds(const unsigned* As, int row0, int rs, int lane) {
  const unsigned* p = As + (row0 + (lane & 15)) * rs + ((lane >> 4) << 3);
  FragU U; U.u[0] = *(const v4u*)p; U.u[1] = *(const v4u*)(p + 4);
  return U.h;
}
// B fragment 32x16 f16 (KxN) from LDS stored [n][kpair] (rs dwords per n-row).
// Lane L holds column N=L&15; lanes 0-15 K=0..15, lanes 16-31 K=16..31.
__device__ __forceinline__ v16h frag_b_lds(const unsigned* Bs, int n0, int rs, int lane) {
  const unsigned* p = Bs + (n0 + (lane & 15)) * rs + ((lane >> 4) << 3);
  FragU U; U.u[0] = *(const v4u*)p; U.u[1] = *(const v4u*)(p + 4);
  return U.h;
}

// ---------------- KV projection: kv = BN(x @ Wkv^T); split -> K[b,h,tok,32], V^T[b,h,d,tok]
__global__ __launch_bounds__(256)
void kv_gemm_kernel(const float* __restrict__ x, const float* __restrict__ Wkv,
                    const float* __restrict__ mean, const float* __restrict__ var,
                    const float* __restrict__ wgt, const float* __restrict__ bia,
                    _Float16* __restrict__ kf, _Float16* __restrict__ vT) {
  __shared__ __align__(16) unsigned As[128 * 16];
  __shared__ __align__(16) unsigned Bs[64 * 16];
  const int tid = threadIdx.x, lane = tid & 31, wv = tid >> 5;
  const int wm = wv >> 1, wn = wv & 1;
  const int m0 = blockIdx.y * 128, n0 = blockIdx.x * 64;
  v8f acc[2][2] = {};
  for (int k0 = 0; k0 < IN_DIMC; k0 += 32) {
    for (int idx = tid; idx < 128 * 8; idx += 256) {
      int row = idx >> 3, q4 = idx & 7;
      v4f f = *(const v4f*)(x + (long)(m0 + row) * IN_DIMC + k0 + q4 * 4);
      As[row * 16 + q4 * 2]     = pack2h(f.x, f.y);
      As[row * 16 + q4 * 2 + 1] = pack2h(f.z, f.w);
    }
    for (int idx = tid; idx < 64 * 8; idx += 256) {
      int row = idx >> 3, q4 = idx & 7;
      v4f f = *(const v4f*)(Wkv + (long)(n0 + row) * IN_DIMC + k0 + q4 * 4);
      Bs[row * 16 + q4 * 2]     = pack2h(f.x, f.y);
      Bs[row * 16 + q4 * 2 + 1] = pack2h(f.z, f.w);
    }
    __syncthreads();
    v16h a0 = frag_a_lds(As, wm * 32,      16, lane);
    v16h a1 = frag_a_lds(As, wm * 32 + 16, 16, lane);
    v16h b0 = frag_b_lds(Bs, wn * 32,      16, lane);
    v16h b1 = frag_b_lds(Bs, wn * 32 + 16, 16, lane);
    acc[0][0] = __builtin_amdgcn_wmma_f32_16x16x32_f16(false, a0, false, b0, (short)0, acc[0][0], false, false);
    acc[0][1] = __builtin_amdgcn_wmma_f32_16x16x32_f16(false, a0, false, b1, (short)0, acc[0][1], false, false);
    acc[1][0] = __builtin_amdgcn_wmma_f32_16x16x32_f16(false, a1, false, b0, (short)0, acc[1][0], false, false);
    acc[1][1] = __builtin_amdgcn_wmma_f32_16x16x32_f16(false, a1, false, b1, (short)0, acc[1][1], false, false);
    __syncthreads();
  }
  for (int i = 0; i < 2; ++i)
    for (int j = 0; j < 2; ++j)
      for (int e = 0; e < 8; ++e) {
        int m = m0 + wm * 32 + i * 16 + ((lane >> 4) << 3) + e;
        int n = n0 + wn * 32 + j * 16 + (lane & 15);
        float y = (acc[i][j][e] - mean[n]) * (wgt[n] * rsqrtf(var[n] + EPSBN)) + bia[n];
        int bb = m / N_TOK, tok = m % N_TOK;
        int hh = n / 96, c = n % 96;
        if (c < KD)
          kf[((long)(bb * H_HEADS + hh) * NKP + tok) * KD + c] = (_Float16)y;
        else
          vT[((long)(bb * H_HEADS + hh) * DVDIM + (c - KD)) * NKP + tok] = (_Float16)y;
      }
}

// ---------------- Q projection on subsampled tokens: q = BN(xs @ Wq^T) -> [b,h,q(pad208),32]
__global__ __launch_bounds__(256)
void q_gemm_kernel(const float* __restrict__ x, const float* __restrict__ Wq,
                   const float* __restrict__ mean, const float* __restrict__ var,
                   const float* __restrict__ wgt, const float* __restrict__ bia,
                   _Float16* __restrict__ qb) {
  __shared__ __align__(16) unsigned As[128 * 16];
  __shared__ __align__(16) unsigned Bs[64 * 16];
  const int tid = threadIdx.x, lane = tid & 31, wv = tid >> 5;
  const int wm = wv >> 1, wn = wv & 1;
  const int m0 = blockIdx.y * 128, n0 = blockIdx.x * 64;
  v8f acc[2][2] = {};
  for (int k0 = 0; k0 < IN_DIMC; k0 += 32) {
    for (int idx = tid; idx < 128 * 8; idx += 256) {
      int row = idx >> 3, q4 = idx & 7;
      int r = m0 + row, bb = r / NQ, t = r % NQ;
      int tok = (t / 14) * 2 * 28 + (t % 14) * 2;   // stride-2 subsample on 28x28 grid
      v4f f = *(const v4f*)(x + ((long)bb * N_TOK + tok) * IN_DIMC + k0 + q4 * 4);
      As[row * 16 + q4 * 2]     = pack2h(f.x, f.y);
      As[row * 16 + q4 * 2 + 1] = pack2h(f.z, f.w);
    }
    for (int idx = tid; idx < 64 * 8; idx += 256) {
      int row = idx >> 3, q4 = idx & 7;
      v4f f = *(const v4f*)(Wq + (long)(n0 + row) * IN_DIMC + k0 + q4 * 4);
      Bs[row * 16 + q4 * 2]     = pack2h(f.x, f.y);
      Bs[row * 16 + q4 * 2 + 1] = pack2h(f.z, f.w);
    }
    __syncthreads();
    v16h a0 = frag_a_lds(As, wm * 32,      16, lane);
    v16h a1 = frag_a_lds(As, wm * 32 + 16, 16, lane);
    v16h b0 = frag_b_lds(Bs, wn * 32,      16, lane);
    v16h b1 = frag_b_lds(Bs, wn * 32 + 16, 16, lane);
    acc[0][0] = __builtin_amdgcn_wmma_f32_16x16x32_f16(false, a0, false, b0, (short)0, acc[0][0], false, false);
    acc[0][1] = __builtin_amdgcn_wmma_f32_16x16x32_f16(false, a0, false, b1, (short)0, acc[0][1], false, false);
    acc[1][0] = __builtin_amdgcn_wmma_f32_16x16x32_f16(false, a1, false, b0, (short)0, acc[1][0], false, false);
    acc[1][1] = __builtin_amdgcn_wmma_f32_16x16x32_f16(false, a1, false, b1, (short)0, acc[1][1], false, false);
    __syncthreads();
  }
  for (int i = 0; i < 2; ++i)
    for (int j = 0; j < 2; ++j)
      for (int e = 0; e < 8; ++e) {
        int m = m0 + wm * 32 + i * 16 + ((lane >> 4) << 3) + e;
        int n = n0 + wn * 32 + j * 16 + (lane & 15);
        float y = (acc[i][j][e] - mean[n]) * (wgt[n] * rsqrtf(var[n] + EPSBN)) + bia[n];
        int bb = m / NQ, t = m % NQ;
        int hh = n / KD, d = n % KD;
        qb[((long)(bb * H_HEADS + hh) * NQ_PAD + t) * KD + d] = (_Float16)y;
      }
}

// ---------------- zero the padded V^T tokens (prob==0 must multiply a finite value)
__global__ __launch_bounds__(256)
void pad_zero_kernel(_Float16* __restrict__ vT) {
  int idx = blockIdx.x * 256 + threadIdx.x;      // B*H*DV*16 total
  int t = idx & 15, dvh = idx >> 4;
  vT[(long)dvh * NKP + N_TOK + t] = (_Float16)0.f;
}

// ---------------- attention: per (b,h,16-query tile): softmax(q k^T * s + bias) @ v, hardswish
__global__ __launch_bounds__(256)
void attn_kernel(const _Float16* __restrict__ qb, const _Float16* __restrict__ kf,
                 const _Float16* __restrict__ vT, const float* __restrict__ biases,
                 const int* __restrict__ idxs, _Float16* __restrict__ hs) {
  __shared__ __align__(16) float    sc[16 * NKP];   // raw scores
  __shared__ __align__(16) _Float16 pr[16 * NKP];   // unnormalized probs (A-frags for AV)
  __shared__ float red[16 * 16];
  __shared__ float rowmax[16];
  __shared__ float rowsum[16];
  __shared__ __align__(16) float avred[4 * 256];
  const int tid = threadIdx.x, lane = tid & 31, wv = tid >> 5;
  const int qt = blockIdx.x;                 // 0..12 (queries padded 196->208)
  const int bh = blockIdx.y;                 // b*16+h
  const int bb = bh >> 4, hh = bh & 15;

  // Q fragment (shared by all key tiles of this workgroup)
  FragU QA;
  { const _Float16* p = qb + ((long)bh * NQ_PAD + qt * 16 + (lane & 15)) * KD + ((lane >> 4) << 4);
    QA.u[0] = *(const v4u*)p; QA.u[1] = *(const v4u*)(p + 8); }

  const _Float16* khead = kf + (long)bh * NKP * KD;
  for (int kt = wv; kt < 50; kt += 8) {      // 50 key tiles of 16 (800 padded keys)
    FragU KB;
    { const _Float16* p = khead + (kt * 16 + (lane & 15)) * KD + ((lane >> 4) << 4);
      KB.u[0] = *(const v4u*)p; KB.u[1] = *(const v4u*)(p + 8); }
    v8f c = {};
    c = __builtin_amdgcn_wmma_f32_16x16x32_f16(false, QA.h, false, KB.h, (short)0, c, false, false);
    for (int e = 0; e < 8; ++e) {
      int m = ((lane >> 4) << 3) + e;
      int key = kt * 16 + (lane & 15);
      float s;
      if (key < N_TOK) {
        int qg = qt * 16 + m; if (qg > NQ - 1) qg = NQ - 1;
        s = c[e] * SCALEQK + biases[hh * 784 + idxs[qg * N_TOK + key]];
      } else s = -1e30f;
      sc[m * NKP + key] = s;
    }
  }
  __syncthreads();

  // softmax: keep unnormalized exp probs + per-row sums (divide in AV epilogue)
  {
    int row = tid >> 4, s16 = tid & 15;
    float mx = -1e30f;
    for (int k = s16; k < NKP; k += 16) mx = fmaxf(mx, sc[row * NKP + k]);
    red[row * 16 + s16] = mx;
    __syncthreads();
    if (s16 == 0) { float m2 = red[row*16]; for (int t = 1; t < 16; ++t) m2 = fmaxf(m2, red[row*16+t]); rowmax[row] = m2; }
    __syncthreads();
    float rm = rowmax[row], sm = 0.f;
    for (int k = s16; k < NKP; k += 16) {
      float e = __expf(sc[row * NKP + k] - rm);
      pr[row * NKP + k] = (_Float16)e;
      sm += e;
    }
    red[row * 16 + s16] = sm;
    __syncthreads();
    if (s16 == 0) { float s2 = 0.f; for (int t = 0; t < 16; ++t) s2 += red[row*16+t]; rowsum[row] = s2; }
  }
  __syncthreads();

  // AV: waves = (dv-tile 0..3) x (K-half 0..1); 25 K-steps of 32
  const int nt = wv & 3, kh = wv >> 2;
  const _Float16* vhead = vT + (long)bh * DVDIM * NKP;
  const unsigned* prU = (const unsigned*)pr;
  v8f accv = {};
  const int ki0 = kh ? 13 : 0, ki1 = kh ? 25 : 13;
  for (int ki = ki0; ki < ki1; ++ki) {
    v16h a = frag_a_lds(prU + ki * 16, 0, NKP / 2, lane);
    FragU VB;
    { const _Float16* p = vhead + (nt * 16 + (lane & 15)) * NKP + ki * 32 + ((lane >> 4) << 4);
      VB.u[0] = *(const v4u*)p; VB.u[1] = *(const v4u*)(p + 8); }
    accv = __builtin_amdgcn_wmma_f32_16x16x32_f16(false, a, false, VB.h, (short)0, accv, false, false);
  }
  if (kh == 1)
    for (int e = 0; e < 8; ++e) avred[nt * 256 + lane * 8 + e] = accv[e];
  __syncthreads();
  if (kh == 0)
    for (int e = 0; e < 8; ++e) {
      float v = accv[e] + avred[nt * 256 + lane * 8 + e];
      int m = ((lane >> 4) << 3) + e;
      int qg = qt * 16 + m;
      if (qg < NQ) {
        float o = v / rowsum[m];
        float r6 = o + 3.f; r6 = r6 < 0.f ? 0.f : (r6 > 6.f ? 6.f : r6);
        hs[((long)bb * NQ + qg) * DHD + hh * DVDIM + nt * 16 + (lane & 15)] = (_Float16)(o * r6 * (1.f / 6.f));
      }
    }
}

// ---------------- output projection: out = BN(hs @ Wp^T) (f32)
__global__ __launch_bounds__(256)
void proj_gemm_kernel(const _Float16* __restrict__ hsb, const float* __restrict__ Wp,
                      const float* __restrict__ mean, const float* __restrict__ var,
                      const float* __restrict__ wgt, const float* __restrict__ bia,
                      float* __restrict__ out) {
  __shared__ __align__(16) unsigned As[128 * 16];
  __shared__ __align__(16) unsigned Bs[64 * 16];
  const int tid = threadIdx.x, lane = tid & 31, wv = tid >> 5;
  const int wm = wv >> 1, wn = wv & 1;
  const int m0 = blockIdx.y * 128, n0 = blockIdx.x * 64;
  const unsigned* hsU = (const unsigned*)hsb;       // 512 dwords per row
  v8f acc[2][2] = {};
  for (int k0 = 0; k0 < DHD; k0 += 32) {
    for (int idx = tid; idx < 128 * 16; idx += 256) {
      int row = idx >> 4, dq = idx & 15;
      As[row * 16 + dq] = hsU[(long)(m0 + row) * (DHD / 2) + (k0 >> 1) + dq];
    }
    for (int idx = tid; idx < 64 * 8; idx += 256) {
      int row = idx >> 3, q4 = idx & 7;
      v4f f = *(const v4f*)(Wp + (long)(n0 + row) * DHD + k0 + q4 * 4);
      Bs[row * 16 + q4 * 2]     = pack2h(f.x, f.y);
      Bs[row * 16 + q4 * 2 + 1] = pack2h(f.z, f.w);
    }
    __syncthreads();
    v16h a0 = frag_a_lds(As, wm * 32,      16, lane);
    v16h a1 = frag_a_lds(As, wm * 32 + 16, 16, lane);
    v16h b0 = frag_b_lds(Bs, wn * 32,      16, lane);
    v16h b1 = frag_b_lds(Bs, wn * 32 + 16, 16, lane);
    acc[0][0] = __builtin_amdgcn_wmma_f32_16x16x32_f16(false, a0, false, b0, (short)0, acc[0][0], false, false);
    acc[0][1] = __builtin_amdgcn_wmma_f32_16x16x32_f16(false, a0, false, b1, (short)0, acc[0][1], false, false);
    acc[1][0] = __builtin_amdgcn_wmma_f32_16x16x32_f16(false, a1, false, b0, (short)0, acc[1][0], false, false);
    acc[1][1] = __builtin_amdgcn_wmma_f32_16x16x32_f16(false, a1, false, b1, (short)0, acc[1][1], false, false);
    __syncthreads();
  }
  for (int i = 0; i < 2; ++i)
    for (int j = 0; j < 2; ++j)
      for (int e = 0; e < 8; ++e) {
        int m = m0 + wm * 32 + i * 16 + ((lane >> 4) << 3) + e;
        int n = n0 + wn * 32 + j * 16 + (lane & 15);
        float y = (acc[i][j][e] - mean[n]) * (wgt[n] * rsqrtf(var[n] + EPSBN)) + bia[n];
        out[(long)m * OUTD + n] = y;
      }
}

extern "C" void kernel_launch(void* const* d_in, const int* in_sizes, int n_in,
                              void* d_out, int out_size, void* d_ws, size_t ws_size,
                              hipStream_t stream) {
  const float* x       = (const float*)d_in[0];
  const float* Wkv     = (const float*)d_in[1];
  const float* kv_mean = (const float*)d_in[2];
  const float* kv_var  = (const float*)d_in[3];
  const float* kv_w    = (const float*)d_in[4];
  const float* kv_b    = (const float*)d_in[5];
  const float* Wq      = (const float*)d_in[6];
  const float* q_mean  = (const float*)d_in[7];
  const float* q_var   = (const float*)d_in[8];
  const float* q_w     = (const float*)d_in[9];
  const float* q_b     = (const float*)d_in[10];
  const float* Wp      = (const float*)d_in[11];
  const float* p_mean  = (const float*)d_in[12];
  const float* p_var   = (const float*)d_in[13];
  const float* p_w     = (const float*)d_in[14];
  const float* p_b     = (const float*)d_in[15];
  const float* biases  = (const float*)d_in[16];
  const int*   idxs    = (const int*)d_in[17];
  (void)in_sizes; (void)n_in; (void)out_size; (void)ws_size;

  // workspace bump allocator (~98 MB total)
  char* ws = (char*)d_ws;
  size_t off = 0;
  auto alloc = [&](size_t bytes) -> void* {
    off = (off + 255) & ~(size_t)255;
    void* p = ws + off; off += bytes; return p;
  };
  _Float16* kf = (_Float16*)alloc((size_t)B_SZ * H_HEADS * NKP * KD * 2);     // 26.2 MB
  _Float16* vT = (_Float16*)alloc((size_t)B_SZ * H_HEADS * DVDIM * NKP * 2);  // 52.4 MB
  _Float16* qb = (_Float16*)alloc((size_t)B_SZ * H_HEADS * NQ_PAD * KD * 2);  //  6.8 MB
  _Float16* hs = (_Float16*)alloc((size_t)B_SZ * NQ * DHD * 2);               // 12.8 MB

  pad_zero_kernel<<<dim3((B_SZ * H_HEADS * DVDIM * 16) / 256), dim3(256), 0, stream>>>(vT);
  kv_gemm_kernel<<<dim3(HIDC / 64, (B_SZ * N_TOK) / 128), dim3(256), 0, stream>>>(
      x, Wkv, kv_mean, kv_var, kv_w, kv_b, kf, vT);
  q_gemm_kernel<<<dim3(NHKD / 64, (B_SZ * NQ) / 128), dim3(256), 0, stream>>>(
      x, Wq, q_mean, q_var, q_w, q_b, qb);
  attn_kernel<<<dim3(13, B_SZ * H_HEADS), dim3(256), 0, stream>>>(
      qb, kf, vT, biases, idxs, hs);
  proj_gemm_kernel<<<dim3(OUTD / 64, (B_SZ * NQ) / 128), dim3(256), 0, stream>>>(
      hs, Wp, p_mean, p_var, p_w, p_b, (float*)d_out);
}
